// DeepSpeedEncoder_12343736008767
// MI455X (gfx1250) — compile-verified, matrix operand order checked by reading, not run on hardware
//
#include <hip/hip_runtime.h>
#include <hip/hip_bf16.h>
#include <math.h>

// ---------------------------------------------------------------------------
// Types for CDNA5 WMMA (wave32): V_WMMA_F32_16X16X32_BF16
// ---------------------------------------------------------------------------
typedef __attribute__((ext_vector_type(16))) __bf16 v16bf;
typedef __attribute__((ext_vector_type(8)))  __bf16 bf16x8;
typedef __attribute__((ext_vector_type(8)))  float  f32x8;

static constexpr int Bc  = 2;
static constexpr int Sc  = 2048;
static constexpr int Hc  = 1024;
static constexpr int NHc = 16;
static constexpr int HDc = 64;     // head dim
static constexpr int DFFc = 4096;
static constexpr int ROWS = Bc * Sc;   // 4096 token rows

__device__ __forceinline__ f32x8 wmma_bf16(v16bf a, v16bf b, f32x8 c) {
  // 8 args: (neg_a, A, neg_b, B, c_mod, C, reuse_a, reuse_b)
  return __builtin_amdgcn_wmma_f32_16x16x32_bf16(false, a, false, b,
                                                 (short)0, c, false, false);
}

// CDNA5 async global->LDS copy (GLOBAL_LOAD_ASYNC_TO_LDS_B128, GV mode).
// Each lane copies 16 bytes; completion tracked with ASYNCcnt.
__device__ __forceinline__ void async_copy_b128(void* lds_dst, const void* gsrc) {
  const unsigned lds = (unsigned)(size_t)lds_dst;               // low 32b = LDS addr
  const unsigned long long ga = (unsigned long long)(size_t)gsrc;
  asm volatile("global_load_async_to_lds_b128 %0, %1, off"
               :: "v"(lds), "v"(ga)
               : "memory");
}
__device__ __forceinline__ void wait_async0() {
  asm volatile("s_wait_asynccnt 0x0" ::: "memory");
}

// ---------------------------------------------------------------------------
// 16-lane row reductions via DPP16 row_ror rotations (pure VALU, no LDS).
// Log-step rotation reduce: after ror 1,2,4,8 every lane holds the full
// 16-lane result. row_ror:n encodes as 0x120+n.
// ---------------------------------------------------------------------------
template <int CTRL>
__device__ __forceinline__ float dpp_f32(float x) {
  const int xi = __float_as_int(x);
  return __int_as_float(
      __builtin_amdgcn_update_dpp(xi, xi, CTRL, 0xF, 0xF, false));
}
__device__ __forceinline__ float rowmax16(float v) {
  v = fmaxf(v, dpp_f32<0x121>(v));
  v = fmaxf(v, dpp_f32<0x122>(v));
  v = fmaxf(v, dpp_f32<0x124>(v));
  v = fmaxf(v, dpp_f32<0x128>(v));
  return v;
}
__device__ __forceinline__ float rowsum16(float v) {
  v += dpp_f32<0x121>(v);
  v += dpp_f32<0x122>(v);
  v += dpp_f32<0x124>(v);
  v += dpp_f32<0x128>(v);
  return v;
}

// Load a 16x32 bf16 fragment (A-layout, also valid for B when the source is
// stored N-major) from a row-major tile with leading dimension `ld` elements.
// ISA layout: lane L -> row = L&15, per-lane K = kb+(0..7) and kb+16+(0..7),
// kb = (L>>4)*8.  Both K runs are contiguous -> two 16B loads.
__device__ __forceinline__ v16bf load_frag16(const __bf16* __restrict__ src, int ld) {
  const int lane = threadIdx.x & 31;
  const int m  = lane & 15;
  const int kb = (lane >> 4) << 3;
  const bf16x8 lo = *(const bf16x8*)(src + (size_t)m * ld + kb);
  const bf16x8 hi = *(const bf16x8*)(src + (size_t)m * ld + kb + 16);
  v16bf f;
#pragma unroll
  for (int j = 0; j < 8; ++j) { f[j] = lo[j]; f[j + 8] = hi[j]; }
  return f;
}

// ---------------------------------------------------------------------------
// Elementwise f32 -> bf16 weight conversion
// ---------------------------------------------------------------------------
__global__ void cvt_f32_to_bf16(const float* __restrict__ in,
                                __bf16* __restrict__ out, int n) {
  int i = blockIdx.x * 256 + threadIdx.x;
  if (i < n) out[i] = (__bf16)in[i];
}

// ---------------------------------------------------------------------------
// LayerNorm (one block of 256 threads per row of H=1024) -> bf16 output
// ---------------------------------------------------------------------------
__global__ __launch_bounds__(256)
void layernorm_to_bf16(const float* __restrict__ x,
                       const float* __restrict__ w,
                       const float* __restrict__ b,
                       __bf16* __restrict__ out, int Hdim) {
  const int row = blockIdx.x;
  const float* xr = x + (size_t)row * Hdim;
  float sum = 0.f, sq = 0.f;
  for (int i = threadIdx.x; i < Hdim; i += 256) {
    float v = xr[i]; sum += v; sq += v * v;
  }
  // wave reduce: 16-lane DPP reduce, then combine halves via shfl
  sum = rowsum16(sum); sq = rowsum16(sq);
  sum += __shfl_xor(sum, 16, 32);
  sq  += __shfl_xor(sq,  16, 32);
  __shared__ float red[2][8];
  if ((threadIdx.x & 31) == 0) {
    red[0][threadIdx.x >> 5] = sum;
    red[1][threadIdx.x >> 5] = sq;
  }
  __syncthreads();
  sum = red[0][threadIdx.x & 7];
  sq  = red[1][threadIdx.x & 7];
#pragma unroll
  for (int off = 4; off; off >>= 1) {
    sum += __shfl_xor(sum, off, 8);
    sq  += __shfl_xor(sq,  off, 8);
  }
  const float mu   = sum / (float)Hdim;
  const float var  = sq / (float)Hdim - mu * mu;
  const float rstd = rsqrtf(var + 1e-12f);
  for (int i = threadIdx.x; i < Hdim; i += 256)
    out[(size_t)row * Hdim + i] = (__bf16)((xr[i] - mu) * rstd * w[i] + b[i]);
}

// ---------------------------------------------------------------------------
// Tiled bf16 WMMA GEMM:  C[M,N] = A[M,K] * W[K,N] + bias (+ epilogue)
// Block tile 128x128, K-step 32, 256 threads = 8 waves in a 4x2 wave grid;
// each wave owns a 32x64 sub-tile = 2x4 WMMA 16x16 accumulators.
// Double-buffered LDS: next tile staged (A via async global->LDS DMA,
// W transposed via VGPRs) while WMMAs run on the current tile.
// ---------------------------------------------------------------------------
enum { EPI_F32 = 0, EPI_F32_RES = 1, EPI_BF16 = 2, EPI_BF16_GELU = 3 };

template <int EPI>
__global__ __launch_bounds__(256)
void gemm_bf16_wmma(const __bf16* __restrict__ A,
                    const __bf16* __restrict__ W,
                    const float* __restrict__ bias,
                    const float* __restrict__ resid,
                    void* __restrict__ Cout,
                    int M, int N, int K) {
  constexpr int BM = 128, BN = 128, BK = 32, LDT = BK + 8; // pad to 40
  __shared__ __bf16 sA[2][BM * LDT];      // A tile, row-major [m][k]
  __shared__ __bf16 sBt[2][BN * LDT];     // W tile, transposed [n][k]

  const int tid  = threadIdx.x;
  const int lane = tid & 31;
  const int wave = tid >> 5;
  const int wm   = wave >> 1;            // 0..3
  const int wn   = wave & 1;             // 0..1
  const int bm0  = blockIdx.y * BM;
  const int bn0  = blockIdx.x * BN;

  auto stage = [&](int buf, int k0) {
#pragma unroll
    for (int it = 0; it < 2; ++it) {
      const int idx = (tid + it * 256) * 8;      // flat bf16 index in tile
      const int r = idx >> 5;                    // 0..127
      const int c = idx & 31;                    // multiple of 8
      async_copy_b128(&sA[buf][r * LDT + c],
                      A + (size_t)(bm0 + r) * K + k0 + c);
      __builtin_prefetch(A + (size_t)(bm0 + r) * K + k0 + BK + c, 0, 1);
    }
#pragma unroll
    for (int it = 0; it < 2; ++it) {
      const int idx = (tid + it * 256) * 8;
      const int kk = idx >> 7;                   // 0..31
      const int nn = idx & 127;                  // multiple of 8
      const bf16x8 v = *(const bf16x8*)(W + (size_t)(k0 + kk) * N + bn0 + nn);
#pragma unroll
      for (int e = 0; e < 8; ++e) sBt[buf][(nn + e) * LDT + kk] = v[e];
      __builtin_prefetch(W + (size_t)(k0 + BK + kk) * N + bn0 + nn, 0, 1);
    }
  };

  f32x8 acc[2][4];
#pragma unroll
  for (int i = 0; i < 2; ++i)
#pragma unroll
    for (int j = 0; j < 4; ++j) acc[i][j] = {};

  stage(0, 0);
  wait_async0();
  __syncthreads();

  int cur = 0;
  for (int k0 = 0; k0 < K; k0 += BK) {
    if (k0 + BK < K) stage(cur ^ 1, k0 + BK);   // prefetch next tile

    v16bf af[2], bfrag[4];
#pragma unroll
    for (int i = 0; i < 2; ++i)
      af[i] = load_frag16(&sA[cur][(wm * 32 + i * 16) * LDT], LDT);
#pragma unroll
    for (int j = 0; j < 4; ++j)
      bfrag[j] = load_frag16(&sBt[cur][(wn * 64 + j * 16) * LDT], LDT);

#pragma unroll
    for (int i = 0; i < 2; ++i)
#pragma unroll
      for (int j = 0; j < 4; ++j)
        acc[i][j] = wmma_bf16(af[i], bfrag[j], acc[i][j]);

    wait_async0();       // next tile's async copies done
    __syncthreads();     // and everyone finished reading `cur`
    cur ^= 1;
  }

  // ---- epilogue: C layout lane = col (lane&15), VGPR r = row r + 8*(lane>>4)
  const int lr  = lane & 15;
  const int hi4 = lane >> 4;
#pragma unroll
  for (int i = 0; i < 2; ++i) {
#pragma unroll
    for (int j = 0; j < 4; ++j) {
      const int col = bn0 + wn * 64 + j * 16 + lr;
      const float bb = bias[col];
#pragma unroll
      for (int r = 0; r < 8; ++r) {
        const int row = bm0 + wm * 32 + i * 16 + r + hi4 * 8;
        const size_t o = (size_t)row * N + col;
        float v = acc[i][j][r] + bb;
        if constexpr (EPI == EPI_F32_RES) {
          ((float*)Cout)[o] = v + resid[o];
        } else if constexpr (EPI == EPI_BF16) {
          ((__bf16*)Cout)[o] = (__bf16)v;
        } else if constexpr (EPI == EPI_BF16_GELU) {
          const float g = 0.5f * v * (1.0f + erff(v * 0.70710678118f));
          ((__bf16*)Cout)[o] = (__bf16)g;
        } else {
          ((float*)Cout)[o] = v;
        }
      }
    }
  }
}

// ---------------------------------------------------------------------------
// Fused streaming attention (flash-style, online softmax), all-WMMA.
// qkv: bf16 [B*S][3H]  (q @ col h*64, k @ col H + h*64, v @ col 2H + h*64)
// grid: (S/64, B*NH); 128 threads = 4 waves, each wave owns 16 query rows.
// K/V tiles staged once per block into double-buffered LDS (K async, V
// transposed via VGPRs); softmax row reductions are DPP16 rotations.
// ---------------------------------------------------------------------------
__global__ __launch_bounds__(128)
void attn_flash_wmma(const __bf16* __restrict__ qkv,
                     const float* __restrict__ mask,      // [B*S]
                     __bf16* __restrict__ ctx_out,        // [B*S][H]
                     int S) {
  constexpr int ld3 = 3 * Hc;
  constexpr int LDK = 72;   // K tile [32 tok][64 d], padded ld
  constexpr int LDV = 40;   // Vt tile [64 d][32 tok], padded ld
  __shared__ __bf16 sK[2][32 * LDK];
  __shared__ __bf16 sVt[2][64 * LDV];
  __shared__ __bf16 pstage_all[4][16 * 40];   // per-wave P re-layout patch

  const int wave = threadIdx.x >> 5;
  const int lane = threadIdx.x & 31;
  const int lr   = lane & 15;
  const int hi4  = lane >> 4;

  const int bh = blockIdx.y;
  const int b  = bh >> 4;           // / NHc
  const int h  = bh & 15;
  const int q0 = blockIdx.x * 64 + wave * 16;
  const int rowbase = b * S;

  __bf16* pstage = &pstage_all[wave][0];

  // staging coordinates: 128 threads cover 32 tokens x 64 dims, 16 elems each
  const int s_tok = threadIdx.x >> 2;         // 0..31
  const int s_d0  = (threadIdx.x & 3) << 4;   // 0,16,32,48

  auto stageKV = [&](int buf, int kv0) {
    const size_t trow = (size_t)(rowbase + kv0 + s_tok) * ld3;
    const __bf16* kg = qkv + trow + Hc + h * HDc + s_d0;
    const __bf16* vg = qkv + trow + 2 * Hc + h * HDc + s_d0;
    // K: straight async copies into row-major [tok][d]
    async_copy_b128(&sK[buf][s_tok * LDK + s_d0], kg);
    async_copy_b128(&sK[buf][s_tok * LDK + s_d0 + 8], kg + 8);
    // V: transpose into [d][tok] through VGPRs
    const bf16x8 v0 = *(const bf16x8*)vg;
    const bf16x8 v1 = *(const bf16x8*)(vg + 8);
#pragma unroll
    for (int e = 0; e < 8; ++e) {
      sVt[buf][(s_d0 + e) * LDV + s_tok] = v0[e];
      sVt[buf][(s_d0 + 8 + e) * LDV + s_tok] = v1[e];
    }
    __builtin_prefetch(kg + 32 * ld3, 0, 1);
    __builtin_prefetch(vg + 32 * ld3, 0, 1);
  };

  // Q fragments: 16 rows x 64 (two K-chunks of 32)
  const __bf16* qbase = qkv + (size_t)(rowbase + q0) * ld3 + h * HDc;
  v16bf qf[2];
  qf[0] = load_frag16(qbase, ld3);
  qf[1] = load_frag16(qbase + 32, ld3);

  float mstate[8], lstate[8];
  f32x8 cacc[4];
#pragma unroll
  for (int r = 0; r < 8; ++r) { mstate[r] = -1e30f; lstate[r] = 0.f; }
#pragma unroll
  for (int t = 0; t < 4; ++t) cacc[t] = {};

  stageKV(0, 0);
  wait_async0();
  __syncthreads();

  int cur = 0;
  for (int kv0 = 0; kv0 < S; kv0 += 32) {
    if (kv0 + 32 < S) stageKV(cur ^ 1, kv0 + 32);  // prefetch next K/V tile

    // ---- scores: two 16x16 tiles over 32 keys (B-frags from sK) ----
    f32x8 sc[2];
#pragma unroll
    for (int j = 0; j < 2; ++j) {
      const __bf16* kbase = &sK[cur][(j * 16) * LDK];
      const v16bf k0f = load_frag16(kbase, LDK);        // kdim 0..31
      const v16bf k1f = load_frag16(kbase + 32, LDK);   // kdim 32..63
      f32x8 z = {};
      z = wmma_bf16(qf[0], k0f, z);
      z = wmma_bf16(qf[1], k1f, z);
      const float mk = mask[rowbase + kv0 + j * 16 + lr];
#pragma unroll
      for (int r = 0; r < 8; ++r) sc[j][r] = z[r] * 0.125f + mk; // 1/sqrt(64)
    }

    // ---- online softmax per row (16-lane DPP reductions, pure VALU) ----
#pragma unroll
    for (int r = 0; r < 8; ++r) {
      const float tmax  = rowmax16(fmaxf(sc[0][r], sc[1][r]));
      const float mnew  = fmaxf(mstate[r], tmax);
      const float alpha = __expf(mstate[r] - mnew);
      const float p0 = __expf(sc[0][r] - mnew);
      const float p1 = __expf(sc[1][r] - mnew);
      sc[0][r] = p0; sc[1][r] = p1;
      const float ps = rowsum16(p0 + p1);
      lstate[r] = lstate[r] * alpha + ps;
      mstate[r] = mnew;
#pragma unroll
      for (int t = 0; t < 4; ++t) cacc[t][r] *= alpha;
    }

    // ---- re-layout P (C-frag -> A-frag) through per-wave LDS patch ----
#pragma unroll
    for (int j = 0; j < 2; ++j)
#pragma unroll
      for (int r = 0; r < 8; ++r)
        pstage[(r + hi4 * 8) * 40 + j * 16 + lr] = (__bf16)sc[j][r];
    // DS ops from one wave are in-order; compiler inserts s_wait_dscnt.
    const v16bf pf = load_frag16(pstage, 40);

    // ---- ctx += P(16x32) x V(32x64): B-frags from transposed sVt ----
#pragma unroll
    for (int t = 0; t < 4; ++t) {
      const v16bf vf = load_frag16(&sVt[cur][(t * 16) * LDV], LDV);
      cacc[t] = wmma_bf16(pf, vf, cacc[t]);
    }

    wait_async0();       // next tile's async copies done
    __syncthreads();     // and everyone finished reading `cur`
    cur ^= 1;
  }

  // ---- finalize: divide by row sums, store bf16 ctx ----
#pragma unroll
  for (int r = 0; r < 8; ++r) {
    const float inv = 1.0f / lstate[r];
#pragma unroll
    for (int t = 0; t < 4; ++t) cacc[t][r] *= inv;
  }
#pragma unroll
  for (int t = 0; t < 4; ++t)
#pragma unroll
    for (int r = 0; r < 8; ++r)
      ctx_out[(size_t)(rowbase + q0 + r + hi4 * 8) * Hc + h * HDc + t * 16 + lr] =
          (__bf16)cacc[t][r];
}

// ---------------------------------------------------------------------------
// Host-side orchestration
// ---------------------------------------------------------------------------
extern "C" void kernel_launch(void* const* d_in, const int* in_sizes, int n_in,
                              void* d_out, int out_size, void* d_ws, size_t ws_size,
                              hipStream_t stream) {
  const float* x       = (const float*)d_in[0];
  const float* mask    = (const float*)d_in[1];
  const float* qkvw    = (const float*)d_in[2];
  const float* qkvb    = (const float*)d_in[3];
  const float* ow      = (const float*)d_in[4];
  const float* ob      = (const float*)d_in[5];
  const float* inter_w = (const float*)d_in[6];
  const float* inter_b = (const float*)d_in[7];
  const float* out_w   = (const float*)d_in[8];
  const float* out_b   = (const float*)d_in[9];
  const float* norm_w  = (const float*)d_in[10];
  const float* norm_b  = (const float*)d_in[11];
  const float* attn_nw = (const float*)d_in[12];
  const float* attn_nb = (const float*)d_in[13];
  float* out = (float*)d_out;

  // ---- workspace carve-out ----
  char* ws = (char*)d_ws;
  size_t off = 0;
  auto carve = [&](size_t bytes) -> void* {
    void* p = ws + off;
    off += (bytes + 255) & ~(size_t)255;
    return p;
  };
  __bf16* h_ln      = (__bf16*)carve((size_t)ROWS * Hc * 2);
  __bf16* qkvw_bf   = (__bf16*)carve((size_t)Hc * 3 * Hc * 2);
  __bf16* ow_bf     = (__bf16*)carve((size_t)Hc * Hc * 2);
  __bf16* interw_bf = (__bf16*)carve((size_t)Hc * DFFc * 2);
  __bf16* outw_bf   = (__bf16*)carve((size_t)DFFc * Hc * 2);
  __bf16* qkv_bf    = (__bf16*)carve((size_t)ROWS * 3 * Hc * 2);
  __bf16* ctx_bf    = (__bf16*)carve((size_t)ROWS * Hc * 2);
  float*  attn_out  = (float*) carve((size_t)ROWS * Hc * 4);
  __bf16* h2_bf     = (__bf16*)carve((size_t)ROWS * Hc * 2);
  __bf16* inter_bf  = (__bf16*)carve((size_t)ROWS * DFFc * 2);

  // ---- weight conversions to bf16 ----
  auto cvt = [&](const float* src, __bf16* dst, int n) {
    cvt_f32_to_bf16<<<(n + 255) / 256, 256, 0, stream>>>(src, dst, n);
  };
  cvt(qkvw,    qkvw_bf,   Hc * 3 * Hc);
  cvt(ow,      ow_bf,     Hc * Hc);
  cvt(inter_w, interw_bf, Hc * DFFc);
  cvt(out_w,   outw_bf,   DFFc * Hc);

  // ---- 1) pre-LN ----
  layernorm_to_bf16<<<ROWS, 256, 0, stream>>>(x, norm_w, norm_b, h_ln, Hc);

  // ---- 2) QKV GEMM: [4096x1024] x [1024x3072] -> bf16 qkv ----
  {
    dim3 g(3 * Hc / 128, ROWS / 128);
    gemm_bf16_wmma<EPI_BF16><<<g, 256, 0, stream>>>(
        h_ln, qkvw_bf, qkvb, nullptr, qkv_bf, ROWS, 3 * Hc, Hc);
  }

  // ---- 3) flash attention ----
  {
    dim3 g(Sc / 64, Bc * NHc);
    attn_flash_wmma<<<g, 128, 0, stream>>>(qkv_bf, mask, ctx_bf, Sc);
  }

  // ---- 4) output projection + residual(x) -> f32 attn_out ----
  {
    dim3 g(Hc / 128, ROWS / 128);
    gemm_bf16_wmma<EPI_F32_RES><<<g, 256, 0, stream>>>(
        ctx_bf, ow_bf, ob, x, attn_out, ROWS, Hc, Hc);
  }

  // ---- 5) post-attention LN ----
  layernorm_to_bf16<<<ROWS, 256, 0, stream>>>(attn_out, attn_nw, attn_nb, h2_bf, Hc);

  // ---- 6) FFN up + exact GeLU -> bf16 inter ----
  {
    dim3 g(DFFc / 128, ROWS / 128);
    gemm_bf16_wmma<EPI_BF16_GELU><<<g, 256, 0, stream>>>(
        h2_bf, interw_bf, inter_b, nullptr, inter_bf, ROWS, DFFc, Hc);
  }

  // ---- 7) FFN down + residual(attn_out) -> f32 out ----
  {
    dim3 g(Hc / 128, ROWS / 128);
    gemm_bf16_wmma<EPI_F32_RES><<<g, 256, 0, stream>>>(
        inter_bf, outw_bf, out_b, attn_out, out, ROWS, Hc, DFFc);
  }
}
